// RN_61108794688044
// MI455X (gfx1250) — matrix operand order repaired and checked by reading, not compiled, not run on hardware
//
#include <hip/hip_runtime.h>
#include <hip/hip_bf16.h>

// ---------- CDNA5 WMMA types ----------
typedef __attribute__((ext_vector_type(16))) __bf16 v16bf;
typedef __attribute__((ext_vector_type(8)))  __bf16 v8bf;
typedef __attribute__((ext_vector_type(8)))  float  v8f;

union BF16x16 { v16bf v; v8bf h[2]; };

// =====================================================================
// Conv (3x3, stride 2, pad 1) + bias + ReLU, direct form. OC fixed = 24.
// =====================================================================
__global__ void conv_s2_relu(const float* __restrict__ in,
                             const float* __restrict__ w,
                             const float* __restrict__ bias,
                             float* __restrict__ out,
                             int Cin, int Hin, int Hout) {
    int idx = blockIdx.x * blockDim.x + threadIdx.x;
    int total = 64 * 24 * Hout * Hout;
    if (idx >= total) return;
    int ow = idx % Hout; int t = idx / Hout;
    int oh = t % Hout;   t /= Hout;
    int oc = t % 24;     int b = t / 24;
    float acc = bias[oc];
    for (int ic = 0; ic < Cin; ++ic) {
        for (int kh = 0; kh < 3; ++kh) {
            int ih = oh * 2 - 1 + kh;
            if (ih < 0 || ih >= Hin) continue;
            for (int kw = 0; kw < 3; ++kw) {
                int iw = ow * 2 - 1 + kw;
                if (iw < 0 || iw >= Hin) continue;
                acc += in[((size_t)(b * Cin + ic) * Hin + ih) * Hin + iw] *
                       w[((oc * Cin + ic) * 3 + kh) * 3 + kw];
            }
        }
    }
    out[idx] = acc > 0.f ? acc : 0.f;   // ReLU fused (BN comes after ReLU in ref)
}

// =====================================================================
// BatchNorm (training stats over B,H,W): one block per channel.
// =====================================================================
__global__ void bn_stats(const float* __restrict__ x,
                         const float* __restrict__ gamma,
                         const float* __restrict__ beta,
                         float* __restrict__ ss,   // [0..23]=scale, [24..47]=shift
                         int HW) {
    const int c = blockIdx.x;          // 24 channels
    const int n = 64 * HW;
    float s = 0.f, s2 = 0.f;
    for (int i = threadIdx.x; i < n; i += blockDim.x) {
        int b = i / HW, r = i - b * HW;
        float v = x[((size_t)(b * 24 + c)) * HW + r];
        s += v; s2 += v * v;
    }
    __shared__ float rs[256], rq[256];
    rs[threadIdx.x] = s; rq[threadIdx.x] = s2;
    __syncthreads();
    for (int off = 128; off > 0; off >>= 1) {
        if (threadIdx.x < off) {
            rs[threadIdx.x] += rs[threadIdx.x + off];
            rq[threadIdx.x] += rq[threadIdx.x + off];
        }
        __syncthreads();
    }
    if (threadIdx.x == 0) {
        float mean = rs[0] / (float)n;
        float var  = rq[0] / (float)n - mean * mean;   // biased, torch-style
        float sc   = gamma[c] * rsqrtf(var + 1e-5f);
        ss[c]      = sc;
        ss[24 + c] = beta[c] - mean * sc;
    }
}

__global__ void bn_apply(float* __restrict__ x, const float* __restrict__ ss,
                         int total, int HW) {
    int i = blockIdx.x * blockDim.x + threadIdx.x;
    if (i >= total) return;
    int c = (i / HW) % 24;
    x[i] = x[i] * ss[c] + ss[24 + c];
}

// =====================================================================
// Objects: xobj[b][n][0..23] = act4[b][c][n]; [24]=n/8.0; [25]=n%8
// =====================================================================
__global__ void build_objects(const float* __restrict__ x4, float* __restrict__ xobj) {
    int idx = blockIdx.x * blockDim.x + threadIdx.x;
    if (idx >= 64 * 64 * 26) return;
    int c = idx % 26; int t = idx / 26;
    int n = t % 64;   int b = t / 64;
    float v;
    if (c < 24)       v = x4[(size_t)(b * 24 + c) * 64 + n];
    else if (c == 24) v = (float)n * 0.125f;     // idx / D (true div)
    else              v = (float)(n & 7);        // idx % D
    xobj[idx] = v;
}

// =====================================================================
// Pack g-weights into bf16 WMMA B-fragment tiles.
// Tile = (kt,nt) 32x16 slab; within tile: elem[lane*16+half] where
//   K = kt*32 + (lane>>4)*8 + (half&7) + (half>>3)*16,  N = nt*16 + (lane&15)
// =====================================================================
__global__ void pack_weights(const float* __restrict__ w0,
                             const float* __restrict__ w1,
                             const float* __restrict__ w2,
                             const float* __restrict__ w3,
                             __bf16* __restrict__ Wp) {
    int idx = blockIdx.x * blockDim.x + threadIdx.x;
    if (idx >= 16384 + 3 * 65536) return;
    float v;
    if (idx < 16384) {
        int e = idx & 511, tile = idx >> 9;
        int nt = tile & 15, kt = tile >> 4;
        int lane = e >> 4, half = e & 15;
        int k = kt * 32 + (lane >> 4) * 8 + (half & 7) + (half >> 3) * 16;
        int n = nt * 16 + (lane & 15);
        v = (k < 63) ? w0[n * 63 + k] : 0.f;     // w0 is [256,63] (out,in)
    } else {
        int r = idx - 16384;
        int layer = r >> 16; r &= 65535;
        const float* w = (layer == 0) ? w1 : (layer == 1) ? w2 : w3;
        int e = r & 511, tile = r >> 9;
        int nt = tile & 15, kt = tile >> 4;
        int lane = e >> 4, half = e & 15;
        int k = kt * 32 + (lane >> 4) * 8 + (half & 7) + (half >> 3) * 16;
        int n = nt * 16 + (lane & 15);
        v = w[n * 256 + k];                      // w is [256,256] (out,in)
    }
    Wp[idx] = (__bf16)v;
}

// =====================================================================
// Fused g_theta. Block = 128 thr = 4 waves. Each wave owns 32 pair-rows
// (TWO 16-row M-tiles) + a private 32x256 bf16 LDS tile (16 KB; 64 KB/blk).
// Two M-tiles per wave => each B fragment (2x global_load_b128) feeds TWO
// v_wmma_f32_16x16x32_bf16, halving L2 weight traffic per FLOP.
// =====================================================================
__global__ void __launch_bounds__(128)
g_theta_wmma(const float* __restrict__ xobj,   // [64][64][26]
             const float* __restrict__ q,      // [64][11]
             const __bf16* __restrict__ Wp,    // packed weights
             const float* __restrict__ gb0, const float* __restrict__ gb1,
             const float* __restrict__ gb2, const float* __restrict__ gb3,
             float* __restrict__ gsum) {       // [64][256], pre-zeroed
    alignas(16) __shared__ __bf16 hbuf[4][32][256];   // 64 KB, 16 KB/wave
    const int tid  = threadIdx.x;
    const int wave = tid >> 5;
    const int lane = tid & 31;
    const int rowBase = (blockIdx.x * 4 + wave) * 32;  // 32 consecutive pair rows
    const int bidx = rowBase >> 12;                    // batch (4096 pairs/batch)
    __bf16 (*h)[256] = hbuf[wave];

    __builtin_prefetch((const void*)(Wp + (size_t)lane * 64), 0, 3);

    // ---- synthesize pair features into LDS: 32 rows x 64 feats (63 + pad) ----
    for (int idx = lane; idx < 32 * 64; idx += 32) {
        int rl = idx >> 6, c = idx & 63;
        int p  = (rowBase + rl) & 4095;
        int j = p >> 6, i = p & 63;
        float v;
        if (c < 26)      v = xobj[(bidx * 64 + i) * 26 + c];
        else if (c < 52) v = xobj[(bidx * 64 + j) * 26 + (c - 26)];
        else if (c < 63) v = q[bidx * 11 + (c - 52)];
        else             v = 0.f;
        h[rl][c] = (__bf16)v;
    }
    __syncthreads();

    const int arow = lane & 15;          // A-matrix row per lane
    const int kgrp = (lane >> 4) * 8;    // K sub-group (0 / 8)
    const int ccol = (lane & 15);        // C/D column within N-tile
    const int crow = (lane >> 4) * 8;    // C/D row offset (0 / 8)

    // ---------------- layer 0: K = 64 (2 frags x 2 M-tiles) ----------------
    BF16x16 A0[2][2];                    // [mtile][kstep]
    #pragma unroll
    for (int m = 0; m < 2; ++m)
        #pragma unroll
        for (int ks = 0; ks < 2; ++ks) { // load A fully into regs first
            A0[m][ks].h[0] = *(const v8bf*)&h[m * 16 + arow][ks * 32 + kgrp];
            A0[m][ks].h[1] = *(const v8bf*)&h[m * 16 + arow][ks * 32 + kgrp + 16];
        }
    for (int nt = 0; nt < 16; ++nt) {
        float bias = gb0[nt * 16 + ccol];
        v8f acc0 = {bias, bias, bias, bias, bias, bias, bias, bias};
        v8f acc1 = acc0;
        #pragma unroll
        for (int ks = 0; ks < 2; ++ks) {
            BF16x16 Bf;
            const v8bf* bp = (const v8bf*)&Wp[(size_t)((ks * 16 + nt) * 512 + lane * 16)];
            Bf.h[0] = bp[0]; Bf.h[1] = bp[1];
            acc0 = __builtin_amdgcn_wmma_f32_16x16x32_bf16(
                false, A0[0][ks].v, false, Bf.v, (short)0, acc0, false, false);
            acc1 = __builtin_amdgcn_wmma_f32_16x16x32_bf16(
                false, A0[1][ks].v, false, Bf.v, (short)0, acc1, false, false);
        }
        #pragma unroll
        for (int r = 0; r < 8; ++r) {    // A regs live -> safe in-place write
            float x0 = acc0[r] > 0.f ? acc0[r] : 0.f;
            float x1 = acc1[r] > 0.f ? acc1[r] : 0.f;
            h[crow + r][nt * 16 + ccol]      = (__bf16)x0;
            h[16 + crow + r][nt * 16 + ccol] = (__bf16)x1;
        }
    }
    __syncthreads();

    // ---------------- layers 1..3: K = 256 (8 frags x 2 M-tiles) ----------------
    for (int layer = 0; layer < 3; ++layer) {
        const __bf16* W = Wp + 16384 + (size_t)layer * 65536;
        const float* gbL = (layer == 0) ? gb1 : (layer == 1) ? gb2 : gb3;
        const bool last = (layer == 2);

        BF16x16 A[2][8];
        #pragma unroll
        for (int m = 0; m < 2; ++m)
            #pragma unroll
            for (int ks = 0; ks < 8; ++ks) {
                A[m][ks].h[0] = *(const v8bf*)&h[m * 16 + arow][ks * 32 + kgrp];
                A[m][ks].h[1] = *(const v8bf*)&h[m * 16 + arow][ks * 32 + kgrp + 16];
            }
        for (int nt = 0; nt < 16; ++nt) {
            float bias = gbL[nt * 16 + ccol];
            v8f acc0 = {bias, bias, bias, bias, bias, bias, bias, bias};
            v8f acc1 = acc0;
            #pragma unroll
            for (int ks = 0; ks < 8; ++ks) {
                BF16x16 Bf;
                const v8bf* bp = (const v8bf*)&W[(size_t)((ks * 16 + nt) * 512 + lane * 16)];
                Bf.h[0] = bp[0]; Bf.h[1] = bp[1];
                acc0 = __builtin_amdgcn_wmma_f32_16x16x32_bf16(
                    false, A[0][ks].v, false, Bf.v, (short)0, acc0, false, false);
                acc1 = __builtin_amdgcn_wmma_f32_16x16x32_bf16(
                    false, A[1][ks].v, false, Bf.v, (short)0, acc1, false, false);
            }
            if (!last) {
                #pragma unroll
                for (int r = 0; r < 8; ++r) {
                    float x0 = acc0[r] > 0.f ? acc0[r] : 0.f;
                    float x1 = acc1[r] > 0.f ? acc1[r] : 0.f;
                    h[crow + r][nt * 16 + ccol]      = (__bf16)x0;
                    h[16 + crow + r][nt * 16 + ccol] = (__bf16)x1;
                }
            } else {
                // ReLU + sum over this wave's 32 rows -> per-batch accumulator
                float s = 0.f;
                #pragma unroll
                for (int r = 0; r < 8; ++r) {
                    s += (acc0[r] > 0.f ? acc0[r] : 0.f);
                    s += (acc1[r] > 0.f ? acc1[r] : 0.f);
                }
                s += __shfl_xor(s, 16, 32);          // fold other half-wave rows
                if (lane < 16)
                    atomicAdd(&gsum[bidx * 256 + nt * 16 + lane], s);
            }
        }
        __syncthreads();
    }
}

// =====================================================================
// f_phi: one block per batch row; LDS-staged dense layers 256->256->256->10
// =====================================================================
__global__ void f_phi(const float* __restrict__ gsum,
                      const float* __restrict__ w0, const float* __restrict__ b0,
                      const float* __restrict__ w1, const float* __restrict__ b1,
                      const float* __restrict__ w2, const float* __restrict__ b2,
                      float* __restrict__ logits) {
    __shared__ float s0[256], s1[256];
    const int b = blockIdx.x, t = threadIdx.x;
    s0[t] = gsum[b * 256 + t];
    __syncthreads();
    float a = b0[t];
    for (int i = 0; i < 256; ++i) a += s0[i] * w0[t * 256 + i];
    s1[t] = a > 0.f ? a : 0.f;
    __syncthreads();
    a = b1[t];
    for (int i = 0; i < 256; ++i) a += s1[i] * w1[t * 256 + i];
    s0[t] = a > 0.f ? a : 0.f;
    __syncthreads();
    if (t < 10) {
        float c = b2[t];
        for (int i = 0; i < 256; ++i) c += s0[i] * w2[t * 256 + i];
        logits[b * 10 + t] = c;
    }
}

// log_softmax over axis=0 (batch dim — faithful to the source)
__global__ void log_softmax_dim0(const float* __restrict__ logits,
                                 float* __restrict__ out) {
    int c = threadIdx.x;
    if (c >= 10) return;
    float m = -3.4e38f;
    for (int b = 0; b < 64; ++b) m = fmaxf(m, logits[b * 10 + c]);
    float s = 0.f;
    for (int b = 0; b < 64; ++b) s += __expf(logits[b * 10 + c] - m);
    float lse = m + __logf(s);
    for (int b = 0; b < 64; ++b) out[b * 10 + c] = logits[b * 10 + c] - lse;
}

// =====================================================================
// Host launch
// =====================================================================
extern "C" void kernel_launch(void* const* d_in, const int* in_sizes, int n_in,
                              void* d_out, int out_size, void* d_ws, size_t ws_size,
                              hipStream_t stream) {
    // Input order: img, q, conv_w[4], conv_b[4], bn_gamma[4], bn_beta[4],
    //              g_w[4], g_b[4], f_w[3], f_b[3]
    const float* img = (const float*)d_in[0];
    const float* q   = (const float*)d_in[1];
    const float* cw[4] = {(const float*)d_in[2], (const float*)d_in[3],
                          (const float*)d_in[4], (const float*)d_in[5]};
    const float* cb[4] = {(const float*)d_in[6], (const float*)d_in[7],
                          (const float*)d_in[8], (const float*)d_in[9]};
    const float* bg[4] = {(const float*)d_in[10], (const float*)d_in[11],
                          (const float*)d_in[12], (const float*)d_in[13]};
    const float* bb[4] = {(const float*)d_in[14], (const float*)d_in[15],
                          (const float*)d_in[16], (const float*)d_in[17]};
    const float* gw[4] = {(const float*)d_in[18], (const float*)d_in[19],
                          (const float*)d_in[20], (const float*)d_in[21]};
    const float* gb[4] = {(const float*)d_in[22], (const float*)d_in[23],
                          (const float*)d_in[24], (const float*)d_in[25]};
    const float* fw[3] = {(const float*)d_in[26], (const float*)d_in[27],
                          (const float*)d_in[28]};
    const float* fb[3] = {(const float*)d_in[29], (const float*)d_in[30],
                          (const float*)d_in[31]};
    float* out = (float*)d_out;

    // ---- workspace carve (all offsets keep 16B alignment) ----
    float* ws = (float*)d_ws;
    float* a1     = ws;             ws += 64 * 24 * 64 * 64;   // 6291456
    float* a2     = ws;             ws += 64 * 24 * 32 * 32;   // 1572864
    float* a3     = ws;             ws += 64 * 24 * 16 * 16;   // 393216
    float* a4     = ws;             ws += 64 * 24 * 8 * 8;     // 98304
    float* xobj   = ws;             ws += 64 * 64 * 26;        // 106496
    float* ss     = ws;             ws += 64;                  // bn scale/shift
    float* gsum   = ws;             ws += 64 * 256;            // 16384
    float* logits = ws;             ws += 640;
    __bf16* Wp    = (__bf16*)ws;    // 212992 bf16

    const int T = 256;
    // ---- conv stack: conv+relu -> stats -> normalize, x4 ----
    {
        int tot = 64 * 24 * 64 * 64;
        conv_s2_relu<<<(tot + T - 1) / T, T, 0, stream>>>(img, cw[0], cb[0], a1, 3, 128, 64);
        bn_stats<<<24, T, 0, stream>>>(a1, bg[0], bb[0], ss, 64 * 64);
        bn_apply<<<(tot + T - 1) / T, T, 0, stream>>>(a1, ss, tot, 64 * 64);
    }
    {
        int tot = 64 * 24 * 32 * 32;
        conv_s2_relu<<<(tot + T - 1) / T, T, 0, stream>>>(a1, cw[1], cb[1], a2, 24, 64, 32);
        bn_stats<<<24, T, 0, stream>>>(a2, bg[1], bb[1], ss, 32 * 32);
        bn_apply<<<(tot + T - 1) / T, T, 0, stream>>>(a2, ss, tot, 32 * 32);
    }
    {
        int tot = 64 * 24 * 16 * 16;
        conv_s2_relu<<<(tot + T - 1) / T, T, 0, stream>>>(a2, cw[2], cb[2], a3, 24, 32, 16);
        bn_stats<<<24, T, 0, stream>>>(a3, bg[2], bb[2], ss, 16 * 16);
        bn_apply<<<(tot + T - 1) / T, T, 0, stream>>>(a3, ss, tot, 16 * 16);
    }
    {
        int tot = 64 * 24 * 8 * 8;
        conv_s2_relu<<<(tot + T - 1) / T, T, 0, stream>>>(a3, cw[3], cb[3], a4, 24, 16, 8);
        bn_stats<<<24, T, 0, stream>>>(a4, bg[3], bb[3], ss, 8 * 8);
        bn_apply<<<(tot + T - 1) / T, T, 0, stream>>>(a4, ss, tot, 8 * 8);
    }

    // ---- objects + packed weights ----
    build_objects<<<(64 * 64 * 26 + T - 1) / T, T, 0, stream>>>(a4, xobj);
    pack_weights<<<(212992 + T - 1) / T, T, 0, stream>>>(gw[0], gw[1], gw[2], gw[3], Wp);

    // ---- g_theta (WMMA) + pair-sum ----
    hipMemsetAsync(gsum, 0, 64 * 256 * sizeof(float), stream);
    g_theta_wmma<<<2048, 128, 0, stream>>>(xobj, q, Wp, gb[0], gb[1], gb[2], gb[3], gsum);

    // ---- f_phi + log_softmax(dim=0) ----
    f_phi<<<64, 256, 0, stream>>>(gsum, fw[0], fb[0], fw[1], fb[1], fw[2], fb[2], logits);
    log_softmax_dim0<<<1, 32, 0, stream>>>(logits, out);
}
